// CAAN_84112639525649
// MI455X (gfx1250) — compile-verified, matrix operand order checked by reading, not run on hardware
//
#include <hip/hip_runtime.h>
#include <hip/hip_bf16.h>

// ---------------------------------------------------------------------------
// CDNA5 / gfx1250, wave32. Matrix math via v_wmma_f32_16x16x32_bf16,
// global->LDS staging via global_load_async_to_lds_b128 (ASYNCcnt),
// WMMA operands read from LDS as whole v16bf (ds_load_b128 pairs) using
// fragment-major swizzled layouts.
// ---------------------------------------------------------------------------

typedef __attribute__((ext_vector_type(16))) __bf16 v16bf;
typedef __attribute__((ext_vector_type(8)))  float  v8f;

#define DEV static __device__ __forceinline__

DEV __bf16 f2bf(float f) {
  unsigned u = __builtin_bit_cast(unsigned, f);
  u += 0x7FFFu + ((u >> 16) & 1u);          // round-to-nearest-even
  unsigned short h = (unsigned short)(u >> 16);
  return __builtin_bit_cast(__bf16, h);
}

// 16-bit A/B-matrix 16x32 fragment layout (ISA 7.12.2):
// VGPR pair p, lane-half hf -> K base
DEV int kmap(int p, int hf) {
  return (p < 4) ? (2 * p + 8 * hf) : (16 + 2 * (p - 4) + 8 * hf);
}
// inverse: local K index (0..31) -> element slot inside v16bf (lane-half is (kl>>3)&1)
DEV int emap(int kl) {
  return 2 * ((((kl >> 4) & 1) << 2) + ((kl & 7) >> 1)) + (kl & 1);
}

// generic LDS pointer -> 32-bit LDS byte offset (generic = {aperture, offset})
DEV unsigned lds_off(const void* p) { return (unsigned)(size_t)p; }

// CDNA5 async DMA: 16 bytes/lane global -> LDS, tracked by ASYNCcnt
DEV void async_ld_b128(unsigned lds_addr, const void* gaddr) {
  asm volatile("global_load_async_to_lds_b128 %0, %1, off"
               :: "v"(lds_addr), "v"(gaddr)
               : "memory");
}
DEV void wait_async0() { asm volatile("s_wait_asynccnt 0" ::: "memory"); }

DEV v8f wmma_bf16(v16bf a, v16bf b, v8f c) {
  return __builtin_amdgcn_wmma_f32_16x16x32_bf16(false, a, false, b,
                                                 (short)0, c, false, false);
}

// ---------------------------------------------------------------------------
// Kernel 1: collapse the O(N^2) pairwise-rank MLP into a 51-entry LUT.
// rel_table[d] = sigmoid( relu(Eemb[d] @ Wr1 + br1) @ Wr2 ),  d in [0,50]
// ---------------------------------------------------------------------------
__global__ void rel_table_kernel(const float* __restrict__ Eemb,
                                 const float* __restrict__ Wr1,
                                 const float* __restrict__ br1,
                                 const float* __restrict__ Wr2,
                                 float* __restrict__ rel) {
  const int d = threadIdx.x;
  if (d > 50) return;
  float emb[32];
#pragma unroll
  for (int e = 0; e < 32; ++e) emb[e] = Eemb[d * 32 + e];
  float acc = 0.f;
  for (int j = 0; j < 16; ++j) {
    float h = br1[j];
#pragma unroll
    for (int e = 0; e < 32; ++e) h += emb[e] * Wr1[e * 16 + j];
    acc += fmaxf(h, 0.f) * Wr2[j];
  }
  rel[d] = 1.f / (1.f + __expf(-acc));
}

// ---------------------------------------------------------------------------
// Kernel 2: C[M,64] = A[M,K] @ W[K,64] + bias.  Weight panel staged once per
// block in LDS, pre-swizzled to fragment order so each lane reads its whole
// v16bf operand as 32 contiguous bytes (2x ds_load_b128).
// Block = 128 threads = 4 waves; each wave owns a 16x64 output tile.
// ---------------------------------------------------------------------------
template <int K>
__global__ void __launch_bounds__(128)
gemm64_bias(const float* __restrict__ A, const float* __restrict__ W,
            const float* __restrict__ bias, float* __restrict__ C) {
  __shared__ __align__(32) __bf16 sW[K / 32][4][32][16];

  const int lane = threadIdx.x & 31;
  const int wave = threadIdx.x >> 5;
  const int hf   = lane >> 4;
  const int l16  = lane & 15;
  const int rowbase = blockIdx.x * 64 + wave * 16;

  // Stage + swizzle the K x 64 weight panel (coalesced global reads).
  for (int idx = threadIdx.x; idx < K * 64; idx += 128) {
    const int kk = idx >> 6, n = idx & 63;
    const int kl = kk & 31;
    const int hfw = (kl >> 3) & 1;
    sW[kk >> 5][n >> 4][(n & 15) | (hfw << 4)][emap(kl)] =
        f2bf(W[(size_t)kk * 64 + n]);
  }
  __syncthreads();

  v8f acc[4] = {};
  const float* arow = A + (size_t)(rowbase + l16) * K;

  for (int k0 = 0; k0 < K; k0 += 32) {
    // A fragment: per lane two contiguous 8-float runs -> wide global loads.
    v16bf a;
#pragma unroll
    for (int p = 0; p < 8; ++p) {
      const int kb = k0 + kmap(p, hf);
      a[2 * p]     = f2bf(arow[kb]);
      a[2 * p + 1] = f2bf(arow[kb + 1]);
    }
#pragma unroll
    for (int c = 0; c < 4; ++c) {
      const v16bf wf = *(const v16bf*)(&sW[k0 >> 5][c][lane][0]);
      acc[c] = wmma_bf16(a, wf, acc[c]);
    }
  }
#pragma unroll
  for (int c = 0; c < 4; ++c) {
    const int n = c * 16 + l16;
    const float bv = bias[n];
#pragma unroll
    for (int r = 0; r < 8; ++r) {
      const int row = rowbase + r + 8 * hf;
      C[(size_t)row * 64 + n] = acc[c][r] + bv;
    }
  }
}

// ---------------------------------------------------------------------------
// Kernel 3: flash attention with rank-distance rel bias from the LUT.
// grid (N/64, B), block 128 = 4 waves, each wave = one 16-row q tile.
// K/V tiles DMA'd global->LDS with async-to-LDS (ASYNCcnt), then swizzled to
// bf16 fragment order; scores never touch HBM; online softmax.
// ---------------------------------------------------------------------------
__global__ void __launch_bounds__(128)
attn_flash(const float* __restrict__ Q, const float* __restrict__ Km,
           const float* __restrict__ V, const int* __restrict__ ranks,
           const float* __restrict__ rel_table, float* __restrict__ AO) {
  __shared__ __align__(16) float  sKraw[32][64];       // 8 KB raw f32 tile
  __shared__ __align__(16) float  sVraw[32][64];       // 8 KB raw f32 tile
  __shared__ __align__(32) __bf16 sKf[2][2][32][16];   // [kchunk][keyhalf][lane][elem]
  __shared__ __align__(32) __bf16 sVf[4][32][16];      // [colchunk][lane][elem]
  __shared__ __align__(32) __bf16 sPf[4][32][16];      // per-wave P in A-frag order

  const int lane = threadIdx.x & 31;
  const int wave = threadIdx.x >> 5;
  const int hf   = lane >> 4;
  const int l16  = lane & 15;
  const int b    = blockIdx.y;
  const int qbase = blockIdx.x * 64 + wave * 16;

  // Hoist q A-fragments (K=64 -> two 32-chunks) for this wave's 16 rows.
  v16bf aq0, aq1;
  {
    const float* arow = Q + (size_t)(b * 512 + qbase + l16) * 64;
#pragma unroll
    for (int p = 0; p < 8; ++p) {
      const int kb = kmap(p, hf);
      aq0[2 * p]     = f2bf(arow[kb]);
      aq0[2 * p + 1] = f2bf(arow[kb + 1]);
      aq1[2 * p]     = f2bf(arow[32 + kb]);
      aq1[2 * p + 1] = f2bf(arow[32 + kb + 1]);
    }
  }
  int rq[8];
#pragma unroll
  for (int r = 0; r < 8; ++r) rq[r] = ranks[b * 512 + qbase + r + 8 * hf];

  float m_run[8], l_run[8];
#pragma unroll
  for (int r = 0; r < 8; ++r) { m_run[r] = -1e30f; l_run[r] = 0.f; }
  v8f o[4] = {};

  const unsigned kraw0 = lds_off(&sKraw[0][0]);
  const unsigned vraw0 = lds_off(&sVraw[0][0]);

  for (int j0 = 0; j0 < 512; j0 += 32) {
    __syncthreads();                       // previous tile fully consumed
    // ---- async DMA: 32 keys x 64 dims of K and V (f32) into LDS ----------
    {
      const float* gk = Km + (size_t)(b * 512 + j0) * 64;
      const float* gv = V  + (size_t)(b * 512 + j0) * 64;
      for (int i = threadIdx.x; i < 512; i += 128) {   // 512 x 16B per tensor
        async_ld_b128(kraw0 + i * 16, gk + i * 4);
        async_ld_b128(vraw0 + i * 16, gv + i * 4);
      }
    }
    wait_async0();
    __syncthreads();                       // raw tiles visible to all waves

    // ---- swizzle: f32 raw -> bf16 fragment-major operands -----------------
    for (int i = threadIdx.x; i < 2048; i += 128) {
      const int key = i >> 6, dim = i & 63;
      const int kl = dim & 31;                       // K-dim local (scores)
      sKf[dim >> 5][key >> 4][(key & 15) | (((kl >> 3) & 1) << 4)][emap(kl)] =
          f2bf(sKraw[key][dim]);
      const int klv = key;                           // K-dim local (P@V)
      sVf[dim >> 4][(dim & 15) | (((klv >> 3) & 1) << 4)][emap(klv)] =
          f2bf(sVraw[key][dim]);
    }
    __syncthreads();

    // ---- scores = q @ k^T for two 16-key sub-tiles (2 WMMAs each) ---------
    v8f s0 = {}, s1 = {};
#pragma unroll
    for (int kc = 0; kc < 2; ++kc) {
      const v16bf b0 = *(const v16bf*)(&sKf[kc][0][lane][0]);
      const v16bf b1 = *(const v16bf*)(&sKf[kc][1][lane][0]);
      const v16bf a = kc ? aq1 : aq0;
      s0 = wmma_bf16(a, b0, s0);
      s1 = wmma_bf16(a, b1, s1);
    }

    const int rk0 = ranks[b * 512 + j0 + l16];
    const int rk1 = ranks[b * 512 + j0 + 16 + l16];
    const int hr  = (l16 >> 3) & 1;                  // A-frag half for this key
    const int el0 = 2 * ((l16 & 7) >> 1) + (l16 & 1);

#pragma unroll
    for (int r = 0; r < 8; ++r) {
      int d0 = rq[r] - rk0; d0 = d0 < 0 ? -d0 : d0; d0 = d0 > 50 ? 50 : d0;
      int d1 = rq[r] - rk1; d1 = d1 < 0 ? -d1 : d1; d1 = d1 > 50 ? 50 : d1;
      const float v0 = s0[r] * 0.125f * rel_table[d0];   // 1/sqrt(64)=0.125
      const float v1 = s1[r] * 0.125f * rel_table[d1];

      // online softmax: row lives across the 16 lanes of this half-wave
      float rowm = fmaxf(v0, v1);
      rowm = fmaxf(rowm, __shfl_xor(rowm, 1));
      rowm = fmaxf(rowm, __shfl_xor(rowm, 2));
      rowm = fmaxf(rowm, __shfl_xor(rowm, 4));
      rowm = fmaxf(rowm, __shfl_xor(rowm, 8));
      const float mnew = fmaxf(m_run[r], rowm);
      const float corr = __expf(m_run[r] - mnew);
      const float p0 = __expf(v0 - mnew);
      const float p1 = __expf(v1 - mnew);
      float rs = p0 + p1;
      rs += __shfl_xor(rs, 1);
      rs += __shfl_xor(rs, 2);
      rs += __shfl_xor(rs, 4);
      rs += __shfl_xor(rs, 8);
      l_run[r] = l_run[r] * corr + rs;
      m_run[r] = mnew;
#pragma unroll
      for (int c = 0; c < 4; ++c) o[c][r] *= corr;

      // store P directly in A-fragment order (same-wave LDS is in-order)
      const int mrow = r + 8 * hf;
      sPf[wave][mrow | (hr << 4)][el0]     = f2bf(p0);
      sPf[wave][mrow | (hr << 4)][8 + el0] = f2bf(p1);
    }

    // ---- o += P(16x32) @ V(32x64) ----------------------------------------
    const v16bf ap = *(const v16bf*)(&sPf[wave][lane][0]);
#pragma unroll
    for (int c = 0; c < 4; ++c) {
      const v16bf bvf = *(const v16bf*)(&sVf[c][lane][0]);
      o[c] = wmma_bf16(ap, bvf, o[c]);
    }
  }

#pragma unroll
  for (int r = 0; r < 8; ++r) {
    const float inv = 1.f / l_run[r];
    const int row = b * 512 + qbase + r + 8 * hf;
#pragma unroll
    for (int c = 0; c < 4; ++c)
      AO[(size_t)row * 64 + c * 16 + l16] = o[c][r] * inv;
  }
}

// ---------------------------------------------------------------------------
// Kernel 4: fused FFN(64->128->64) + LayerNorm + head(64->32->1) + sigmoid.
// One wave per row; shuffle broadcasts for the tiny GEMVs.
// ---------------------------------------------------------------------------
__global__ void __launch_bounds__(128)
ffn_ln_head(const float* __restrict__ AO,
            const float* __restrict__ Wf1, const float* __restrict__ bf1,
            const float* __restrict__ Wf2, const float* __restrict__ bf2,
            const float* __restrict__ ln_g, const float* __restrict__ ln_b,
            const float* __restrict__ Ws1, const float* __restrict__ bs1,
            const float* __restrict__ Ws2, const float* __restrict__ bs2,
            float* __restrict__ out, int M) {
  const int lane = threadIdx.x & 31;
  const int row = blockIdx.x * 4 + (threadIdx.x >> 5);
  if (row >= M) return;
  const float* a = AO + (size_t)row * 64;
  const float a0 = a[lane];
  const float a1 = a[lane + 32];

  float hbuf[4];
#pragma unroll
  for (int t = 0; t < 4; ++t) {
    const int j = lane + 32 * t;
    float s = bf1[j];
    for (int q = 0; q < 32; ++q) {
      s += __shfl(a0, q) * Wf1[q * 128 + j];
      s += __shfl(a1, q) * Wf1[(32 + q) * 128 + j];
    }
    hbuf[t] = fmaxf(s, 0.f);
  }
  float h0 = bf2[lane], h1 = bf2[lane + 32];
#pragma unroll
  for (int t = 0; t < 4; ++t) {
    for (int q = 0; q < 32; ++q) {
      const float hv = __shfl(hbuf[t], q);
      const int j = 32 * t + q;
      h0 += hv * Wf2[j * 64 + lane];
      h1 += hv * Wf2[j * 64 + lane + 32];
    }
  }
  float sum = h0 + h1;
  for (int m = 1; m <= 16; m <<= 1) sum += __shfl_xor(sum, m);
  const float mu = sum * (1.f / 64.f);
  const float d0 = h0 - mu, d1 = h1 - mu;
  float vs = d0 * d0 + d1 * d1;
  for (int m = 1; m <= 16; m <<= 1) vs += __shfl_xor(vs, m);
  const float rstd = rsqrtf(vs * (1.f / 64.f) + 1e-5f);
  const float hn0 = d0 * rstd * ln_g[lane] + ln_b[lane];
  const float hn1 = d1 * rstd * ln_g[lane + 32] + ln_b[lane + 32];

  float s = bs1[lane];
  for (int q = 0; q < 32; ++q) {
    s += __shfl(hn0, q) * Ws1[q * 32 + lane];
    s += __shfl(hn1, q) * Ws1[(32 + q) * 32 + lane];
  }
  s = fmaxf(s, 0.f);
  float oacc = s * Ws2[lane];
  for (int m = 1; m <= 16; m <<= 1) oacc += __shfl_xor(oacc, m);
  if (lane == 0) out[row] = 1.f / (1.f + __expf(-(oacc + bs2[0])));
}

// ---------------------------------------------------------------------------
extern "C" void kernel_launch(void* const* d_in, const int* in_sizes, int n_in,
                              void* d_out, int out_size, void* d_ws, size_t ws_size,
                              hipStream_t stream) {
  (void)in_sizes; (void)n_in; (void)out_size; (void)ws_size;
  const float* x    = (const float*)d_in[0];
  const int*   rnk  = (const int*)d_in[1];
  const float* Wp   = (const float*)d_in[2];
  const float* bp   = (const float*)d_in[3];
  const float* Wq   = (const float*)d_in[4];
  const float* bq   = (const float*)d_in[5];
  const float* Wk   = (const float*)d_in[6];
  const float* bk   = (const float*)d_in[7];
  const float* Wv   = (const float*)d_in[8];
  const float* bv   = (const float*)d_in[9];
  const float* Eemb = (const float*)d_in[10];
  const float* Wr1  = (const float*)d_in[11];
  const float* br1  = (const float*)d_in[12];
  const float* Wr2  = (const float*)d_in[13];
  const float* Wf1  = (const float*)d_in[14];
  const float* bf1  = (const float*)d_in[15];
  const float* Wf2  = (const float*)d_in[16];
  const float* bf2  = (const float*)d_in[17];
  const float* lng  = (const float*)d_in[18];
  const float* lnb  = (const float*)d_in[19];
  const float* Ws1  = (const float*)d_in[20];
  const float* bs1  = (const float*)d_in[21];
  const float* Ws2  = (const float*)d_in[22];
  const float* bs2  = (const float*)d_in[23];

  // Workspace: rel LUT + xp/q/k/v/ao (each 8192x64 f32) = ~10.5 MB (L2-resident)
  float* ws  = (float*)d_ws;
  float* rel = ws;
  float* xp  = ws + 64;
  float* q   = xp + 8192 * 64;
  float* k   = q  + 8192 * 64;
  float* v   = k  + 8192 * 64;
  float* ao  = v  + 8192 * 64;

  rel_table_kernel<<<1, 64, 0, stream>>>(Eemb, Wr1, br1, Wr2, rel);
  gemm64_bias<256><<<128, 128, 0, stream>>>(x,  Wp, bp, xp);
  gemm64_bias<64><<<128, 128, 0, stream>>>(xp, Wq, bq, q);
  gemm64_bias<64><<<128, 128, 0, stream>>>(xp, Wk, bk, k);
  gemm64_bias<64><<<128, 128, 0, stream>>>(xp, Wv, bv, v);
  attn_flash<<<dim3(8, 16), 128, 0, stream>>>(q, k, v, rnk, rel, ao);
  ffn_ln_head<<<2048, 128, 0, stream>>>(ao, Wf1, bf1, Wf2, bf2, lng, lnb,
                                        Ws1, bs1, Ws2, bs2, (float*)d_out, 8192);
}
